// FlowPos2dEmd_13494787244721
// MI455X (gfx1250) — matrix operand order repaired and checked by reading, not compiled, not run on hardware
//
#include <hip/hip_runtime.h>
#include <hip/hip_bf16.h>

// CDNA5 (gfx1250) wave32 WMMA types
typedef __attribute__((ext_vector_type(16))) __bf16 v16bf;
typedef __attribute__((ext_vector_type(8)))  __bf16 v8bf;
typedef __attribute__((ext_vector_type(8)))  float  v8f;

// Convert 8 fp32 -> 8 bf16 (RNE via __bf16 cast)
__device__ __forceinline__ v8bf cvt8(float4 a, float4 b) {
  v8bf r;
  r[0] = (__bf16)a.x; r[1] = (__bf16)a.y; r[2] = (__bf16)a.z; r[3] = (__bf16)a.w;
  r[4] = (__bf16)b.x; r[5] = (__bf16)b.y; r[6] = (__bf16)b.z; r[7] = (__bf16)b.w;
  return r;
}

// fast silu: x * rcp(1 + exp(-x))  -> v_exp_f32 + v_rcp_f32 (approx, ~1 ulp)
__device__ __forceinline__ float fast_silu(float x) {
  return x * __builtin_amdgcn_rcpf(1.0f + __expf(-x));
}

// LDS (dynamic, 320 KB):
//   [0      , 65536 )  bf16 : W1 in WMMA B-operand layout
//   [65536  , 131072)  bf16 : W2 in WMMA B-operand layout
//   [131072 , 163840)  bf16 : per-wave H staging, wave w -> 16x256 bf16 tile
//
// B-operand layout for 16x16x32 bf16 WMMA (32x16 tile, kc = K-chunk, n = N-tile):
//   element ((kc*16+n)*32 + lane)*16 + e  ==  W[kc*32 + (e>>3)*16 + (lane>>4)*8 + (e&7)][n*16 + (lane&15)]
// A-operand layout (16x32 tile): lane = half*16 + m holds row m,
//   elems 0..7  = K = kc*32 + half*8 + e
//   elems 8..15 = K = kc*32 + 16 + half*8 + e

__global__ __launch_bounds__(256) void flowpos2d_mlp_kernel(
    const float* __restrict__ disc,   // (numTok, 256)  - streamed, NT
    const float* __restrict__ flows,  // (numTok, 3)    - streamed, NT
    const float* __restrict__ pos2d,  // (224,224,256)  - gathered, keep in L2 (RT)
    const float* __restrict__ w1,     // (256, 256) row-major [k][n]
    const float* __restrict__ b1,     // (256,)
    const float* __restrict__ w2,     // (256, 256)
    const float* __restrict__ b2,     // (256,)
    float* __restrict__ out,          // (numTok, 256)  - streamed, NT
    int numTok)
{
  extern __shared__ __bf16 lds[];
  __bf16* ldsW1 = lds;            // 65536 bf16 = 128 KB
  __bf16* ldsW2 = lds + 65536;    // 65536 bf16 = 128 KB
  __bf16* ldsH  = lds + 131072;   // 8 waves * 4096 bf16 = 64 KB

  const int tid = threadIdx.x;

  // ---- one-time cooperative weight convert fp32 -> bf16 B-operand layout ----
  // i = k*256 + col  => coalesced global reads
  for (int i = tid; i < 65536; i += 256) {
    const int k   = i >> 8;
    const int col = i & 255;
    const int kc = k >> 5, kr = k & 31;
    const int hh = (kr >> 3) & 1, e = ((kr >> 4) << 3) | (kr & 7);
    const int n  = col >> 4, c = col & 15;
    const int dst = (((kc << 4) + n) * 32 + (hh << 4) + c) * 16 + e;
    ldsW1[dst] = (__bf16)w1[i];
    ldsW2[dst] = (__bf16)w2[i];
  }
  __syncthreads();

  const int wave = tid >> 5;
  const int lane = tid & 31;
  const int hlf  = lane >> 4;   // K-half select for A/B operands
  const int m    = lane & 15;   // A row / D column-within-tile
  __bf16* Hl = ldsH + wave * 4096;  // this wave's 16x256 bf16 H tile

  const int tiles = (numTok + 127) >> 7;   // 128 tokens per WG-tile (16 per wave)
  for (int tile = blockIdx.x; tile < tiles; tile += gridDim.x) {
    const int base = (tile << 7) + (wave << 4);

    // ---- gather: pe row for token (base + m), truncation + clamp like reference ----
    int tok  = base + m;
    int tokC = tok < numTok ? tok : (numTok - 1);
    const float f0 = __builtin_nontemporal_load(flows + tokC * 3 + 0);
    const float f1 = __builtin_nontemporal_load(flows + tokC * 3 + 1);
    int xi = (int)(f0 * 224.0f);
    int yi = (int)(f1 * 224.0f);
    xi = xi < 0 ? 0 : (xi > 223 ? 223 : xi);
    yi = yi < 0 ? 0 : (yi > 223 ? 223 : yi);
    const float* perow = pos2d + (size_t)(yi * 224 + xi) * 256;

    // ---- load PE tile directly into WMMA A-operand registers (bf16) ----
    v16bf a_pe[8];
#pragma unroll
    for (int kc = 0; kc < 8; ++kc) {
      const float4* p0 = (const float4*)(perow + kc * 32 + hlf * 8);
      const float4* p1 = (const float4*)(perow + kc * 32 + 16 + hlf * 8);
      v8bf lo = cvt8(p0[0], p0[1]);
      v8bf hi = cvt8(p1[0], p1[1]);
      a_pe[kc] = __builtin_shufflevector(lo, hi,
                   0,1,2,3,4,5,6,7,8,9,10,11,12,13,14,15);
    }

    // ---- GEMM1: H = silu(PE @ W1 + b1)  ->  per-wave LDS tile (bf16) ----
    for (int n = 0; n < 16; ++n) {
      v8f acc = {};
#pragma unroll
      for (int kc = 0; kc < 8; ++kc) {
        const v16bf bmat =
            *(const v16bf*)(ldsW1 + (((kc << 4) + n) * 32 + lane) * 16);
        acc = __builtin_amdgcn_wmma_f32_16x16x32_bf16(
                  false, a_pe[kc], false, bmat, (short)0, acc, false, false);
      }
      const int col = (n << 4) + m;            // D-layout: N = n*16 + (lane&15)
      const float bias = b1[col];
#pragma unroll
      for (int r = 0; r < 8; ++r) {            // D-layout: M = r + 8*(lane>=16)
        Hl[(r + hlf * 8) * 256 + col] = (__bf16)fast_silu(acc[r] + bias);
      }
    }

    // same-wave DS ops are in-order; make the cross-lane H exchange explicit
    asm volatile("s_wait_dscnt 0" ::: "memory");

    // ---- reload H transposed into A-operand registers ----
    v16bf a_h[8];
#pragma unroll
    for (int kc = 0; kc < 8; ++kc) {
      const v8bf* p0 = (const v8bf*)(Hl + m * 256 + kc * 32 + hlf * 8);
      const v8bf* p1 = (const v8bf*)(Hl + m * 256 + kc * 32 + 16 + hlf * 8);
      a_h[kc] = __builtin_shufflevector(p0[0], p1[0],
                  0,1,2,3,4,5,6,7,8,9,10,11,12,13,14,15);
    }

    // ---- GEMM2: out = disc + H @ W2 + b2  (streamed NT to protect L2) ----
    for (int n = 0; n < 16; ++n) {
      v8f acc = {};
#pragma unroll
      for (int kc = 0; kc < 8; ++kc) {
        const v16bf bmat =
            *(const v16bf*)(ldsW2 + (((kc << 4) + n) * 32 + lane) * 16);
        acc = __builtin_amdgcn_wmma_f32_16x16x32_bf16(
                  false, a_h[kc], false, bmat, (short)0, acc, false, false);
      }
      const int col = (n << 4) + m;
      const float bias = b2[col];
#pragma unroll
      for (int r = 0; r < 8; ++r) {
        const int row = r + hlf * 8;
        const int t2  = base + row;
        if (t2 < numTok) {
          const size_t idx = (size_t)t2 * 256 + col;
          const float dv = __builtin_nontemporal_load(disc + idx);
          __builtin_nontemporal_store(dv + acc[r] + bias, out + idx);
        }
      }
    }
  }
}

extern "C" void kernel_launch(void* const* d_in, const int* in_sizes, int n_in,
                              void* d_out, int out_size, void* d_ws, size_t ws_size,
                              hipStream_t stream) {
  (void)n_in; (void)d_ws; (void)ws_size; (void)out_size;
  const float* disc  = (const float*)d_in[0];
  const float* flows = (const float*)d_in[1];
  const float* pos2d = (const float*)d_in[2];
  const float* w1    = (const float*)d_in[3];
  const float* b1    = (const float*)d_in[4];
  const float* w2    = (const float*)d_in[5];
  const float* b2    = (const float*)d_in[6];
  float* out = (float*)d_out;

  const int numTok = in_sizes[0] / 256;           // B*N tokens
  const int tiles  = (numTok + 127) >> 7;         // 128 tokens per workgroup tile
  const int grid   = tiles < 512 ? tiles : 512;   // persistent WGs, grid-stride

  // 128 KB W1 + 128 KB W2 + 8 waves * 8 KB H = 320 KB (CDNA5 WGP LDS max)
  const size_t shmem = (size_t)(65536 + 65536 + 8 * 4096) * sizeof(__bf16);

  flowpos2d_mlp_kernel<<<grid, 256, shmem, stream>>>(
      disc, flows, pos2d, w1, b1, w2, b2, out, numTok);
}
